// DNetwork_89455578841390
// MI455X (gfx1250) — matrix operand order repaired and checked by reading, not compiled, not run on hardware
//
#include <hip/hip_runtime.h>

// ---------------------------------------------------------------------------
// Spiking net on MI455X (gfx1250, wave32, WMMA + TDM).
// GEMM1 (dominant, ~1.7e11 bf16 FLOP): 8-wave workgroup computes 128Mx128N,
// wave -> 4x2 tiles of 16x16 (v_wmma_f32_16x16x32_bf16, f32 accum).
// A/B panels are DMA'd into double-buffered LDS by the Tensor Data Mover
// (tensor_load_to_lds, TENSORcnt / s_wait_tensorcnt), so the copy of panel
// it+1 overlaps the 8-WMMA compute of panel it.
// ---------------------------------------------------------------------------

typedef __attribute__((ext_vector_type(16))) __bf16 v16bf;
typedef __attribute__((ext_vector_type(8)))  float  v8f;
typedef unsigned int __attribute__((ext_vector_type(4))) tdm_g0_t;  // D# group0
typedef int          __attribute__((ext_vector_type(8))) tdm_g1_t;  // D# group1
typedef int          __attribute__((ext_vector_type(4))) tdm_g2_t;  // D# group2/3

#define B_      64
#define T_      350
#define NIN_    784
#define NHID_   800
#define NOUT_   10
#define SHIFTS  6            // integer shifts 0..DMAX+1
#define K1P     800          // NIN padded to multiple of 32
#define M1P     896          // NHID padded to 7*128 for BM=128 blocking
#define TT      352          // S1 time cols (T padded to 16)
#define TN1     384          // GEMM1 N extent (3 blocks of 128)
#define TPAD    8            // front zero pad (> DMAX+1)
#define TROWS   396          // TPAD + TN1 + 4 slack rows (staging window)
#define TP3     360          // X3 rows = TPAD + TT
#define DECAY   0.90483741803595952f   // exp(-1/10)
#define THETA_  10.0f

#define BM      128
#define BN      128
#define BROWS   136          // staged B rows (133 needed: 128 + 5 shift)
#define ITERS   (SHIFTS * (K1P / 32))   // 150 (s,kt) panel steps

// LDS byte offsets of the double-buffered panels (dynamic LDS base = 0,
// no static __shared__ in this kernel).
#define A_BYTES (BM * 32 * 2)           // 8192
#define B_BYTES (BROWS * 32 * 2)        // 8704
#define AOFF(p) ((unsigned)((p) * A_BYTES))
#define BOFF(p) ((unsigned)(2 * A_BYTES + (p) * B_BYTES))
#define LDS_TOTAL (2 * A_BYTES + 2 * B_BYTES)   // 33792 B of 320 KB WGP LDS

// ---- TDM descriptor builders (ISA cdna5 §8) -------------------------------
__device__ __forceinline__ tdm_g0_t tdm_group0(unsigned lds_off,
                                               unsigned long long gaddr) {
  tdm_g0_t g;
  g[0] = 1u;                                              // count=1, user D#
  g[1] = lds_off;                                         // lds_addr (bytes)
  g[2] = (unsigned)(gaddr & 0xFFFFFFFFu);                 // global_addr[31:0]
  g[3] = (unsigned)((gaddr >> 32) & 0x1FFFFFFu)           // global_addr[56:32]
       | (2u << 30);                                      // type = 2 ("image")
  return g;
}

__device__ __forceinline__ tdm_g1_t tdm_group1(unsigned tile0, unsigned tile1,
                                               unsigned long long stride0) {
  // data_size=1 (2 bytes); tensor_dim0/1 = 2^31-1 (padding guarantees bounds);
  // 2-D tile: tile_dim0 x tile_dim1, row stride = stride0 (data_size units).
  tdm_g1_t g;
  g[0] = (int)(1u << 16);                                 // data_size=2B
  g[1] = (int)(0xFFFFu << 16);                            // tensor_dim0[15:0]
  g[2] = (int)(0x7FFFu | (0xFFFFu << 16));                // td0[31:16], td1[15:0]
  g[3] = (int)(0x7FFFu | (tile0 << 16));                  // td1[31:16], tile_dim0
  g[4] = (int)(tile1 & 0xFFFFu);                          // tile_dim1; tile_dim2=0
  g[5] = (int)(stride0 & 0xFFFFFFFFu);                    // stride0[31:0]
  g[6] = (int)((stride0 >> 32) & 0xFFFFu);                // stride0[47:32]
  g[7] = 0;
  return g;
}

__device__ __forceinline__ void tdm_load_2d(unsigned lds_off, const void* gptr,
                                            unsigned tile0, unsigned tile1,
                                            unsigned long long stride_elems) {
  tdm_g0_t g0 = tdm_group0(lds_off, (unsigned long long)gptr);
  tdm_g1_t g1 = tdm_group1(tile0, tile1, stride_elems);
  tdm_g2_t z4 = {};
#if defined(__clang_major__) && (__clang_major__ >= 23)
  tdm_g1_t z8 = {};
  __builtin_amdgcn_tensor_load_to_lds(g0, g1, z4, z4, z8, 0);
#else
  __builtin_amdgcn_tensor_load_to_lds(g0, g1, z4, z4, 0);
#endif
}

// ---- psp over raw input spikes: u[t] = a*u[t-1] + s[t]; bf16 [b][t][k] ----
__global__ void psp_input_kernel(const float* __restrict__ spk, __bf16* __restrict__ X1) {
  int k = blockIdx.x * blockDim.x + threadIdx.x;
  int b = blockIdx.y;
  if (k >= K1P) return;
  __bf16* xb = X1 + (size_t)b * TROWS * K1P + k;
  if (k >= NIN_) {
    for (int t = 0; t < TROWS; ++t) xb[(size_t)t * K1P] = (__bf16)0.0f;
    return;
  }
  for (int t = 0; t < TPAD; ++t) xb[(size_t)t * K1P] = (__bf16)0.0f;
  for (int t = TPAD + T_; t < TROWS; ++t) xb[(size_t)t * K1P] = (__bf16)0.0f;
  const float* s = spk + ((size_t)b * NIN_ + k) * T_;
  float u = 0.0f;
  for (int t = 0; t < T_; ++t) {
    u = DECAY * u + s[t];
    xb[(size_t)(TPAD + t) * K1P] = (__bf16)u;
  }
}

// ---- psp over hidden spikes (bf16 [b][h][TT]) -> bf16 [b][t][h] ----
__global__ void psp_hidden_kernel(const __bf16* __restrict__ S1, __bf16* __restrict__ X3) {
  int h = blockIdx.x * blockDim.x + threadIdx.x;
  int b = blockIdx.y;
  if (h >= NHID_) return;
  __bf16* xb = X3 + (size_t)b * TP3 * NHID_ + h;
  for (int t = 0; t < TPAD; ++t) xb[(size_t)t * NHID_] = (__bf16)0.0f;
  xb[(size_t)(TPAD + T_)     * NHID_] = (__bf16)0.0f;
  xb[(size_t)(TPAD + T_ + 1) * NHID_] = (__bf16)0.0f;
  const __bf16* sp = S1 + ((size_t)b * NHID_ + h) * TT;
  float u = 0.0f;
  for (int t = 0; t < T_; ++t) {
    u = DECAY * u + (float)sp[t];
    xb[(size_t)(TPAD + t) * NHID_] = (__bf16)u;
  }
}

// ---- delay-interp coefficient: (fl==s)*(1-fr) + (fl==s-1)*fr, d in [0,4] ----
__device__ __forceinline__ float shift_coeff(float d, int s) {
  d = fminf(fmaxf(d, 0.0f), 4.0f);
  float fl = floorf(d), fr = d - fl, c = 0.0f;
  if (fl == (float)s)       c += 1.0f - fr;
  if (fl == (float)(s - 1)) c += fr;
  return c;
}

// ---- W1 -> 6 shift-scaled bf16 copies, [s][M1P][K1P] (M,K zero padded) ----
__global__ void pack_w1_kernel(const float* __restrict__ w, const float* __restrict__ d,
                               __bf16* __restrict__ out) {
  int idx = blockIdx.x * blockDim.x + threadIdx.x;
  if (idx >= SHIFTS * M1P * K1P) return;
  int s = idx / (M1P * K1P);
  int r = idx % (M1P * K1P);
  int o = r / K1P, k = r % K1P;
  float v = 0.0f;
  if (o < NHID_ && k < NIN_) v = w[o * NIN_ + k] * shift_coeff(d[o * NIN_ + k], s);
  out[idx] = (__bf16)v;
}

// ---- W3 -> [s][16][800] (M padded 10 -> 16) ----
__global__ void pack_w3_kernel(const float* __restrict__ w, const float* __restrict__ d,
                               __bf16* __restrict__ out) {
  int idx = blockIdx.x * blockDim.x + threadIdx.x;
  if (idx >= SHIFTS * 16 * NHID_) return;
  int s = idx / (16 * NHID_);
  int r = idx % (16 * NHID_);
  int o = r / NHID_, k = r % NHID_;
  float v = 0.0f;
  if (o < NOUT_) v = w[o * NHID_ + k] * shift_coeff(d[o * NHID_ + k], s);
  out[idx] = (__bf16)v;
}

// ---- layer-1 GEMM: TDM double-buffered LDS, 8 waves, 4x2 tiles/wave ----
// A frag (16x32 bf16): lane L holds M=L&15; lanes<16: K 0-7,16-23; lanes>=16: K 8-15,24-31.
// B frag (32x16 bf16): lane L holds N=L&15; lanes<16: K 0-7,16-23; lanes>=16: K 8-15,24-31.
__global__ __launch_bounds__(256) void gemm1_kernel(const __bf16* __restrict__ W,
                                                    const __bf16* __restrict__ X,
                                                    __bf16* __restrict__ S1) {
  extern __shared__ __bf16 smem[];     // [ A0 | A1 | B0 | B1 ] = 33 KB
  const int tid  = threadIdx.x;
  const int lane = tid & 31;
  const int wave = tid >> 5;
  const int wm   = wave >> 2;          // 0..1 : wave's 64-row M group
  const int wn   = wave & 3;           // 0..3 : wave's 32-col N group
  const int half = (lane >= 16);
  const int l16  = lane & 15;
  const int mblk = blockIdx.x * BM;
  const int tblk = blockIdx.y * BN;
  const int b    = blockIdx.z;

  v8f acc[4][2];
  #pragma unroll
  for (int mi = 0; mi < 4; ++mi)
    #pragma unroll
    for (int ni = 0; ni < 2; ++ni) acc[mi][ni] = (v8f){};

  // wave 0 drives the Tensor Data Mover; panels indexed by it = kt*6 + s
  if (wave == 0) {
    const __bf16* ga = W + (size_t)(0 * M1P + mblk) * K1P + 0;
    const __bf16* gb = X + ((size_t)b * TROWS + (TPAD - 5) + tblk) * K1P + 0;
    tdm_load_2d(AOFF(0), ga, 32, BM,    K1P);
    tdm_load_2d(BOFF(0), gb, 32, BROWS, K1P);
  }

  #pragma unroll 1
  for (int it = 0; it < ITERS; ++it) {
    const int cur = it & 1;
    const int s   = it % SHIFTS;
    __syncthreads();                   // everyone done reading buffer cur^1
    if (wave == 0) {
      if (it + 1 < ITERS) {            // DMA next panels while others compute
        const int kt2 = (it + 1) / SHIFTS, s2 = (it + 1) % SHIFTS;
        const __bf16* ga = W + (size_t)(s2 * M1P + mblk) * K1P + kt2 * 32;
        const __bf16* gb = X + ((size_t)b * TROWS + (TPAD - 5) + tblk) * K1P + kt2 * 32;
        tdm_load_2d(AOFF(cur ^ 1), ga, 32, BM,    K1P);
        tdm_load_2d(BOFF(cur ^ 1), gb, 32, BROWS, K1P);
        __builtin_amdgcn_s_wait_tensorcnt(2);   // current pair complete
      } else {
        __builtin_amdgcn_s_wait_tensorcnt(0);
      }
    }
    __syncthreads();                   // publish buffer cur

    const __bf16* A  = smem + (size_t)cur * (A_BYTES / 2);
    const __bf16* Bp = smem + (size_t)(2 * A_BYTES + cur * B_BYTES) / 2;

    v16bf af[4], bfr[2];
    #pragma unroll
    for (int mi = 0; mi < 4; ++mi) {
      int aoff = (wm * 64 + mi * 16 + l16) * 32 + (half ? 8 : 0);
      ((uint4*)&af[mi])[0] = *(const uint4*)&A[aoff];
      ((uint4*)&af[mi])[1] = *(const uint4*)&A[aoff + 16];
    }
    #pragma unroll
    for (int ni = 0; ni < 2; ++ni) {
      int brow = wn * 32 + ni * 16 + l16 + 5 - s;   // shifted view of shared panel
      int boff = brow * 32 + (half ? 8 : 0);
      ((uint4*)&bfr[ni])[0] = *(const uint4*)&Bp[boff];
      ((uint4*)&bfr[ni])[1] = *(const uint4*)&Bp[boff + 16];
    }
    #pragma unroll
    for (int mi = 0; mi < 4; ++mi)
      #pragma unroll
      for (int ni = 0; ni < 2; ++ni)
        acc[mi][ni] = __builtin_amdgcn_wmma_f32_16x16x32_bf16(
            false, af[mi], false, bfr[ni], (short)0, acc[mi][ni], false, false);
  }

  // threshold -> hidden spikes (bf16), C/D layout: VGPR r -> M=r (+8 for lanes>=16)
  #pragma unroll
  for (int mi = 0; mi < 4; ++mi) {
    int obase = mblk + wm * 64 + mi * 16 + (half ? 8 : 0);
    #pragma unroll
    for (int ni = 0; ni < 2; ++ni) {
      int t = tblk + wn * 32 + ni * 16 + l16;
      if (t < TT) {
        #pragma unroll
        for (int r = 0; r < 8; ++r) {
          int o = obase + r;
          if (o < NHID_)
            S1[((size_t)b * NHID_ + o) * TT + t] =
                (__bf16)((acc[mi][ni][r] >= THETA_) ? 1.0f : 0.0f);
        }
      }
    }
  }
}

// ---- layer-2 GEMM (<2% of FLOPs): one wave per 16x16 tile, K=800 ----
__global__ void gemm2_kernel(const __bf16* __restrict__ W, const __bf16* __restrict__ X,
                             float* __restrict__ out) {
  const int lane  = threadIdx.x;
  const int tbase = blockIdx.y * 16;
  const int b     = blockIdx.z;
  const int half  = (lane >= 16);
  const int M     = (lane & 15);
  v8f acc = {};
  for (int s = 0; s < SHIFTS; ++s) {
    const __bf16* wrow = W + (size_t)(s * 16 + M) * NHID_ + (half ? 8 : 0);
    const int     trow = TPAD + tbase + (lane & 15) - s;
    const __bf16* xrow = X + ((size_t)b * TP3 + trow) * NHID_ + (half ? 8 : 0);
    for (int kt = 0; kt < NHID_ / 32; ++kt) {
      const __bf16* wp = wrow + kt * 32;
      const __bf16* xp = xrow + kt * 32;
      v16bf afrag, bfrag;
      ((uint4*)&afrag)[0] = *(const uint4*)(wp);
      ((uint4*)&afrag)[1] = *(const uint4*)(wp + 16);
      ((uint4*)&bfrag)[0] = *(const uint4*)(xp);
      ((uint4*)&bfrag)[1] = *(const uint4*)(xp + 16);
      acc = __builtin_amdgcn_wmma_f32_16x16x32_bf16(
          false, afrag, false, bfrag, (short)0, acc, false, false);
    }
  }
  const int tcol = tbase + (lane & 15);
  #pragma unroll
  for (int r = 0; r < 8; ++r) {
    int o = r + (half ? 8 : 0);
    if (o < NOUT_ && tcol < T_)
      out[((size_t)b * NOUT_ + o) * T_ + tcol] = (acc[r] >= THETA_) ? 1.0f : 0.0f;
  }
}

extern "C" void kernel_launch(void* const* d_in, const int* in_sizes, int n_in,
                              void* d_out, int out_size, void* d_ws, size_t ws_size,
                              hipStream_t stream) {
  (void)in_sizes; (void)n_in; (void)out_size; (void)ws_size;
  const float* spk = (const float*)d_in[0];   // [64,784,350]
  const float* w1  = (const float*)d_in[1];   // [800,784]
  const float* d1  = (const float*)d_in[2];   // [800,784]
  const float* w3  = (const float*)d_in[3];   // [10,800]
  const float* d3  = (const float*)d_in[4];   // [10,800]
  float* out = (float*)d_out;                 // [64,10,350]

  // Workspace partition (all chunks 256B-aligned): ~122 MB total.
  char* ws = (char*)d_ws;
  __bf16* X1  = (__bf16*)ws; ws += (size_t)B_ * TROWS * K1P * 2;       // 40.6 MB
  __bf16* W1S = (__bf16*)ws; ws += (size_t)SHIFTS * M1P * K1P * 2;     //  8.6 MB
  __bf16* S1  = (__bf16*)ws; ws += (size_t)B_ * NHID_ * TT * 2;        // 36.0 MB
  __bf16* X3  = (__bf16*)ws; ws += (size_t)B_ * TP3 * NHID_ * 2;       // 36.9 MB
  __bf16* W3S = (__bf16*)ws; ws += (size_t)SHIFTS * 16 * NHID_ * 2;    //  0.15 MB

  psp_input_kernel<<<dim3((K1P + 255) / 256, B_), 256, 0, stream>>>(spk, X1);
  pack_w1_kernel<<<(SHIFTS * M1P * K1P + 255) / 256, 256, 0, stream>>>(w1, d1, W1S);
  pack_w3_kernel<<<(SHIFTS * 16 * NHID_ + 255) / 256, 256, 0, stream>>>(w3, d3, W3S);
  gemm1_kernel<<<dim3(M1P / BM, TN1 / BN, B_), 256, LDS_TOTAL, stream>>>(W1S, X1, S1);
  psp_hidden_kernel<<<dim3((NHID_ + 255) / 256, B_), 256, 0, stream>>>(S1, X3);
  gemm2_kernel<<<dim3(1, TT / 16, B_), 32, 0, stream>>>(W3S, X3, out);
}
